// Decoder_61495341744280
// MI455X (gfx1250) — matrix operand order repaired and checked by reading, not compiled, not run on hardware
//
#include <hip/hip_runtime.h>

// Problem constants (from reference)
#define T_IN    128
#define BATCH   4096
#define HDIM    64
#define T_OUT   24
#define OVERLAP 8
#define POS_DIM 4
#define G3H     192   // 3*H

// Kernel config
#define NB       16    // batches per workgroup
#define NTHREADS 256   // 8 wave32s
#define ROWP     33    // u32 (bf16x2) per encoder row: 32 data + 1 pad (odd -> conflict-free)

typedef __attribute__((ext_vector_type(16))) __bf16        v16bf;
typedef __attribute__((ext_vector_type(16))) unsigned short ushort16;
typedef __attribute__((ext_vector_type(8)))  float          v8f;

// ---- dynamic LDS layout (bytes) ----
#define OFF_ENC   0
#define SZ_ENC    (NB*T_IN*ROWP*4)            // 270336: encoder slice, bf16x2 packed
#define OFF_HID   (OFF_ENC  + SZ_ENC)         // f32 [NB][64]  hidden state
#define OFF_HARR  (OFF_HID  + NB*HDIM*4)      // f32 [NB][64]  h = hidden + context
#define OFF_HABF  (OFF_HARR + NB*HDIM*4)      // u32 [NB][32]  h as bf16x2 (WMMA A staging)
#define OFF_LAST  (OFF_HABF + NB*(HDIM/2)*4)  // f32 [NB][8]
#define OFF_POS   (OFF_LAST + NB*OVERLAP*4)   // f32 [24][4]
#define OFF_WIH   (OFF_POS  + T_OUT*POS_DIM*4)// f32 [192][12]
#define OFF_BIH   (OFF_WIH  + G3H*12*4)       // f32 [192]
#define OFF_BHH   (OFF_BIH  + G3H*4)          // f32 [192]
#define OFF_GI    (OFF_BHH  + G3H*4)          // f32 [NB][192]
#define OFF_GH    (OFF_GI   + NB*G3H*4)       // f32 [NB][192]
#define LDS_BYTES (OFF_GH   + NB*G3H*4)       // = 316800 <= 320KB

// ---- bf16 helpers (RNE) ----
__device__ __forceinline__ unsigned int bf16r(float x) {
  unsigned int u = __float_as_uint(x);
  return (u + 0x7fffu + ((u >> 16) & 1u)) >> 16;
}
__device__ __forceinline__ unsigned int packbf2(float lo, float hi) {
  return bf16r(lo) | (bf16r(hi) << 16);
}
__device__ __forceinline__ float bflo(unsigned int u) { return __uint_as_float(u << 16); }
__device__ __forceinline__ float bfhi(unsigned int u) { return __uint_as_float(u & 0xffff0000u); }

// ---- fast transcendentals: v_rcp_f32 / v_exp_f32, no IEEE div sequences ----
__device__ __forceinline__ float fast_rcp(float x) { return __builtin_amdgcn_rcpf(x); }
__device__ __forceinline__ float fast_sigmoid(float x) {
  return fast_rcp(1.0f + __expf(-x));
}
__device__ __forceinline__ float fast_tanh(float x) {
  // tanh(x) = 2*sigmoid(2x) - 1
  return fmaf(2.0f, fast_sigmoid(2.0f * x), -1.0f);
}

// ---- WMMA fragment gathers, per CDNA5 ISA 7.12.2 layouts (wave32) ----
// A (16x32 bf16, M=batch, K): lane<16 holds K {0..7,16..23}, lane>=16 holds K {8..15,24..31}
__device__ __forceinline__ v16bf load_a_frag(const unsigned int* habf, int lane, int kc) {
  ushort16 u;
  const int m  = lane & 15;
  const int hi = lane >> 4;
#pragma unroll
  for (int e = 0; e < 16; ++e) {
    int k = (e < 8 ? e : e + 8) + (hi ? 8 : 0) + 32 * kc;
    unsigned int w = habf[m * (HDIM / 2) + (k >> 1)];
    u[e] = (unsigned short)((k & 1) ? (w >> 16) : (w & 0xffffu));
  }
  return __builtin_bit_cast(v16bf, u);
}
// B (32x16 bf16, K x N=j): lane<16 -> N=lane, K {0..7,16..23}; lane>=16 -> N=lane-16, K {8..15,24..31}
__device__ __forceinline__ v16bf load_b_frag(const float* __restrict__ W_hh, int jt, int kc, int lane) {
  ushort16 u;
  const int n  = lane & 15;
  const int hi = lane >> 4;
#pragma unroll
  for (int e = 0; e < 16; ++e) {
    int k = (e < 8 ? e : e + 8) + (hi ? 8 : 0) + 32 * kc;
    u[e] = (unsigned short)bf16r(W_hh[(jt * 16 + n) * HDIM + k]);
  }
  return __builtin_bit_cast(v16bf, u);
}

__global__ void __launch_bounds__(NTHREADS)
decoder_attn_gru_kernel(const float* __restrict__ enc,      // (T_IN, B, H)
                        const float* __restrict__ enc_hid,  // (B, H)
                        const float* __restrict__ last,     // (B, OVERLAP)
                        const float* __restrict__ pos,      // (T_OUT, POS_DIM)
                        const float* __restrict__ W_ih,     // (192, 12)
                        const float* __restrict__ W_hh,     // (192, 64)
                        const float* __restrict__ b_ih,     // (192)
                        const float* __restrict__ b_hh,     // (192)
                        const float* __restrict__ W_lin,    // (64)
                        const float* __restrict__ b_lin,    // (1)
                        float* __restrict__ out)            // (B, T_OUT)
{
  extern __shared__ __align__(16) unsigned char smem[];
  unsigned int* encB = (unsigned int*)(smem + OFF_ENC);
  float*        hid  = (float*)(smem + OFF_HID);
  float*        harr = (float*)(smem + OFF_HARR);
  unsigned int* habf = (unsigned int*)(smem + OFF_HABF);
  float*        lsb  = (float*)(smem + OFF_LAST);
  float*        posE = (float*)(smem + OFF_POS);
  float*        Wih  = (float*)(smem + OFF_WIH);
  float*        bih  = (float*)(smem + OFF_BIH);
  float*        bhh  = (float*)(smem + OFF_BHH);
  float*        gi   = (float*)(smem + OFF_GI);
  float*        gh   = (float*)(smem + OFF_GH);

  const int tid   = threadIdx.x;
  const int lane  = tid & 31;
  const int wv    = tid >> 5;                                        // wave id (vector)
  const int wvu   = __builtin_amdgcn_readfirstlane((int)(tid >> 5)); // wave id (scalar/uniform)
  const int bbase = blockIdx.x * NB;

  // ---- Phase 0: stage encoder slice to LDS as bf16x2, coalesced float4 reads ----
  for (int i = tid; i < NB * T_IN * 16; i += NTHREADS) {
    const int row = i >> 4, q = i & 15;
    const int b = row >> 7, t = row & 127;
    const float4 v = ((const float4*)(enc + ((size_t)t * BATCH + bbase + b) * HDIM))[q];
    unsigned int* dst = encB + (b * T_IN + t) * ROWP + q * 2;
    dst[0] = packbf2(v.x, v.y);
    dst[1] = packbf2(v.z, v.w);
  }
  for (int i = tid; i < NB * HDIM; i += NTHREADS)
    hid[i] = enc_hid[(size_t)(bbase + (i >> 6)) * HDIM + (i & 63)];
  for (int i = tid; i < NB * OVERLAP; i += NTHREADS)
    lsb[i] = last[(size_t)(bbase + (i >> 3)) * OVERLAP + (i & 7)];
  if (tid < T_OUT * POS_DIM) posE[tid] = pos[tid];
  for (int i = tid; i < G3H * 12; i += NTHREADS) Wih[i] = W_ih[i];
  if (tid < G3H) { bih[tid] = b_ih[tid]; bhh[tid] = b_hh[tid]; }

  // W_lin / b_lin live in registers (lane l owns h-dims l and l+32)
  const float wl0 = W_lin[lane];
  const float wl1 = W_lin[lane + 32];
  const float blin = b_lin[0];

  // W_hh B-fragments: loop-invariant, keep in VGPRs. Wave w owns j-tiles {w, w+8 (if <12)}.
  const int jt2 = (wvu < 4) ? (wvu + 8) : wvu; // dummy alias for waves without 2nd tile
  v16bf fB00 = load_b_frag(W_hh, wvu, 0, lane);
  v16bf fB01 = load_b_frag(W_hh, wvu, 1, lane);
  v16bf fB10 = load_b_frag(W_hh, jt2, 0, lane);
  v16bf fB11 = load_b_frag(W_hh, jt2, 1, lane);

  __syncthreads();

  // ---- 24 sequential decoder steps ----
  for (int s = 0; s < T_OUT; ++s) {
    // Phase A: additive attention + context; each wave handles 2 batches.
    for (int q = 0; q < 2; ++q) {
      const int b = wv * 2 + q;
      const float* hb = hid + b * HDIM;
      // logits over 128 timesteps, 4 per lane
      float lg[4];
#pragma unroll
      for (int tt = 0; tt < 4; ++tt) {
        const unsigned int* row = encB + (b * T_IN + tt * 32 + lane) * ROWP;
        float acc = 0.f;
#pragma unroll
        for (int p = 0; p < 32; ++p) {
          unsigned int u = row[p];
          acc = fmaf(bflo(u), hb[2 * p], acc);
          acc = fmaf(bfhi(u), hb[2 * p + 1], acc);
        }
        lg[tt] = acc;
      }
      // softmax over time (wave reduction)
      float m = fmaxf(fmaxf(lg[0], lg[1]), fmaxf(lg[2], lg[3]));
#pragma unroll
      for (int off = 16; off; off >>= 1) m = fmaxf(m, __shfl_xor(m, off));
      float aw[4];
      float ssum = 0.f;
#pragma unroll
      for (int tt = 0; tt < 4; ++tt) { aw[tt] = __expf(lg[tt] - m); ssum += aw[tt]; }
#pragma unroll
      for (int off = 16; off; off >>= 1) ssum += __shfl_xor(ssum, off);
      const float inv = fast_rcp(ssum);
#pragma unroll
      for (int tt = 0; tt < 4; ++tt) aw[tt] *= inv;
      // context: lane owns h-dims 2*lane, 2*lane+1; attention weights via lane broadcast
      float c0 = 0.f, c1 = 0.f;
#pragma unroll
      for (int tt = 0; tt < 4; ++tt) {
        const float wsrc = aw[tt];
        for (int i = 0; i < 32; ++i) {
          const float wgt = __shfl(wsrc, i);
          unsigned int u = encB[(b * T_IN + tt * 32 + i) * ROWP + lane];
          c0 = fmaf(bflo(u), wgt, c0);
          c1 = fmaf(bfhi(u), wgt, c1);
        }
      }
      const float h0 = hb[2 * lane] + c0;
      const float h1 = hb[2 * lane + 1] + c1;
      harr[b * HDIM + 2 * lane]     = h0;
      harr[b * HDIM + 2 * lane + 1] = h1;
      habf[b * (HDIM / 2) + lane]   = packbf2(h0, h1);
    }
    __syncthreads();

    // Phase B1: gi = [last, pos_s] @ W_ih.T + b_ih  (K=12, VALU)
    for (int i = tid; i < NB * G3H; i += NTHREADS) {
      const int b = i / G3H, j = i - b * G3H;
      const float* wr = Wih + j * 12;
      float acc = bih[j];
#pragma unroll
      for (int k = 0; k < OVERLAP; ++k) acc = fmaf(lsb[b * OVERLAP + k], wr[k], acc);
#pragma unroll
      for (int k = 0; k < POS_DIM; ++k) acc = fmaf(posE[s * POS_DIM + k], wr[OVERLAP + k], acc);
      gi[i] = acc;
    }

    // Phase B2: gh = h @ W_hh.T via bf16 WMMA (16 batches x 64) x (64 x 192), 12 tiles
    {
      const v16bf a0 = load_a_frag(habf, lane, 0);
      const v16bf a1 = load_a_frag(habf, lane, 1);
      const int n  = lane & 15;
      const int mo = (lane >> 4) ? 8 : 0;
      {
        v8f acc = {};
        acc = __builtin_amdgcn_wmma_f32_16x16x32_bf16(false, a0, false, fB00, (short)0, acc, false, false);
        acc = __builtin_amdgcn_wmma_f32_16x16x32_bf16(false, a1, false, fB01, (short)0, acc, false, false);
#pragma unroll
        for (int r = 0; r < 8; ++r) gh[(r + mo) * G3H + wvu * 16 + n] = acc[r];
      }
      if (wvu < 4) {  // scalar branch: EXEC stays all-ones around WMMA
        v8f acc = {};
        acc = __builtin_amdgcn_wmma_f32_16x16x32_bf16(false, a0, false, fB10, (short)0, acc, false, false);
        acc = __builtin_amdgcn_wmma_f32_16x16x32_bf16(false, a1, false, fB11, (short)0, acc, false, false);
#pragma unroll
        for (int r = 0; r < 8; ++r) gh[(r + mo) * G3H + (wvu + 8) * 16 + n] = acc[r];
      }
    }
    __syncthreads();

    // Phase C: GRU gates -> new hidden (hardware rcp/exp, no IEEE div sequences)
    for (int i = tid; i < NB * HDIM; i += NTHREADS) {
      const int b = i >> 6, hd = i & 63;
      const float ir = gi[b * G3H + hd];
      const float iz = gi[b * G3H + 64 + hd];
      const float in_ = gi[b * G3H + 128 + hd];
      const float hr = gh[b * G3H + hd] + bhh[hd];
      const float hz = gh[b * G3H + 64 + hd] + bhh[64 + hd];
      const float hn = gh[b * G3H + 128 + hd] + bhh[128 + hd];
      const float r = fast_sigmoid(ir + hr);
      const float z = fast_sigmoid(iz + hz);
      const float nn = fast_tanh(fmaf(r, hn, in_));
      hid[i] = fmaf(1.0f - z, nn, z * harr[i]);
    }
    __syncthreads();

    // Phase D: out = h_new @ W_lin.T + b_lin; feed back into last[:,0]
    for (int q = 0; q < 2; ++q) {
      const int b = wv * 2 + q;
      float part = fmaf(hid[b * HDIM + lane], wl0, hid[b * HDIM + 32 + lane] * wl1);
#pragma unroll
      for (int off = 16; off; off >>= 1) part += __shfl_xor(part, off);
      if (lane == 0) {
        const float o = part + blin;
        out[(size_t)(bbase + b) * T_OUT + s] = o;
        lsb[b * OVERLAP + 0] = o;
      }
    }
    __syncthreads();
  }
}

extern "C" void kernel_launch(void* const* d_in, const int* in_sizes, int n_in,
                              void* d_out, int out_size, void* d_ws, size_t ws_size,
                              hipStream_t stream) {
  (void)in_sizes; (void)n_in; (void)out_size; (void)d_ws; (void)ws_size;
  const float* enc     = (const float*)d_in[0];
  const float* enc_hid = (const float*)d_in[1];
  const float* last    = (const float*)d_in[2];
  const float* pos     = (const float*)d_in[3];
  const float* W_ih    = (const float*)d_in[4];
  const float* W_hh    = (const float*)d_in[5];
  const float* b_ih    = (const float*)d_in[6];
  const float* b_hh    = (const float*)d_in[7];
  const float* W_lin   = (const float*)d_in[8];
  const float* b_lin   = (const float*)d_in[9];
  float* out = (float*)d_out;

  decoder_attn_gru_kernel<<<dim3(BATCH / NB), dim3(NTHREADS), LDS_BYTES, stream>>>(
      enc, enc_hid, last, pos, W_ih, W_hh, b_ih, b_hh, W_lin, b_lin, out);
}
